// GE2E_Loss_32289564131480
// MI455X (gfx1250) — compile-verified, matrix operand order checked by reading, not run on hardware
//
#include <hip/hip_runtime.h>
#include <hip/hip_bf16.h>
#include <math.h>

typedef __attribute__((ext_vector_type(16))) __bf16 v16bf;
typedef __attribute__((ext_vector_type(8)))  __bf16 v8bf;
typedef __attribute__((ext_vector_type(8)))  float  v8f;

#define N_SPK 1024
#define M_UTT 32
#define D_DIM 128
#define NROWS (N_SPK * M_UTT)   // 32768
#define GEEPS 1e-8f

__device__ __forceinline__ float sigmoidf_(float x) { return 1.0f / (1.0f + expf(-x)); }

// Load one WMMA 16-bit operand chunk (K=32) in the ISA bf16 A/B striping:
// lanes 0-15: elems 0..7 = K off..off+7, elems 8..15 = K off+16..off+23
// lanes 16-31 get the +8-shifted halves via `off` (caller adds kbase).
__device__ __forceinline__ v16bf load_mat16(const __bf16* p, int off) {
  v8bf lo = *(const v8bf*)(p + off);
  v8bf hi = *(const v8bf*)(p + off + 16);
  v16bf r;
#pragma unroll
  for (int i = 0; i < 8; i++) { r[i] = lo[i]; r[i + 8] = hi[i]; }
  return r;
}

// Block (128 threads = 4 waves) sum-reduction of two values, broadcast to all.
__device__ __forceinline__ void block_reduce2(float& a, float& b,
                                              volatile float* s0, volatile float* s1) {
#pragma unroll
  for (int o = 16; o > 0; o >>= 1) {
    a += __shfl_xor(a, o, 32);
    b += __shfl_xor(b, o, 32);
  }
  int wid = threadIdx.x >> 5;
  if ((threadIdx.x & 31) == 0) { s0[wid] = a; s1[wid] = b; }
  __syncthreads();
  a = s0[0] + s0[1] + s0[2] + s0[3];
  b = s1[0] + s1[1] + s1[2] + s1[3];
  __syncthreads();
}

// Kernel 1: one block (128 threads) per speaker n.
// Produces: bf16 e_unit rows, bf16 (w * c_unit) rows, per-row pos term (LOO diag, fp32).
__global__ void ge2e_prep(const float* __restrict__ emb,
                          const float* __restrict__ wp,
                          const float* __restrict__ bp,
                          __bf16* __restrict__ eb,
                          __bf16* __restrict__ cb,
                          float*  __restrict__ posval) {
  __shared__ float ldsE[M_UTT][D_DIM];
  __shared__ float s0[4], s1[4];

  const int n = blockIdx.x;
  const int t = threadIdx.x;                       // dim index 0..127
  const float* base = emb + (size_t)n * M_UTT * D_DIM;

  float S = 0.f;                                   // per-dim sum over utterances
#pragma unroll 4
  for (int m = 0; m < M_UTT; m++) {
    float x = base[m * D_DIM + t];
    ldsE[m][t] = x;
    S += x;
  }
  __syncthreads();

  float s2 = S * S, dummy = 0.f;                   // -> ||S||^2
  block_reduce2(s2, dummy, s0, s1);

  const float w = wp[0], bb = bp[0];

  const float normS = sqrtf(s2);
  const float cden  = fmaxf(normS * (1.0f / M_UTT), GEEPS);   // max(||centroid||, eps)
  // Fold w into the B panel: WMMA then directly produces w*cos (sigmoid is monotone
  // in its argument, so max_k sigmoid(w*cos+b) = sigmoid(max_k(w*cos) + b), any sign of w).
  cb[n * D_DIM + t] = (__bf16)(w * (S * (1.0f / M_UTT)) / cden);

  for (int m = 0; m < M_UTT; m++) {
    float x  = ldsE[m][t];
    float e2 = x * x;                              // -> ||e||^2
    float es = x * S;                              // -> e . S
    block_reduce2(e2, es, s0, s1);

    float ne     = sqrtf(e2);
    // loo = (S - e)/(M-1):  e.loo = (e.S - ||e||^2)/(M-1)
    // ||loo||^2 = (||S||^2 - 2 e.S + ||e||^2)/(M-1)^2
    float dotLoo = (es - e2) * (1.0f / (M_UTT - 1));
    float loo2   = fmaxf(s2 - 2.f * es + e2, 0.f) * (1.0f / ((M_UTT - 1) * (M_UTT - 1)));
    float nl     = sqrtf(loo2);
    float cosloo = dotLoo / (fmaxf(ne, GEEPS) * fmaxf(nl, GEEPS));

    eb[(size_t)(n * M_UTT + m) * D_DIM + t] = (__bf16)(x / fmaxf(ne, GEEPS));
    if (t == 0) posval[n * M_UTT + m] = 1.0f - sigmoidf_(w * cosloo + bb);
  }
}

// Kernel 2: fused GEMM + row-max of w*cos with leave-one-out column excluded.
// One wave per 16-row tile; 32 iterations x 2 column tiles, two interleaved WMMA chains.
// Exclusion is branch-free: per-element v_cndmask against an SGPR-foldable mask.
__global__ void ge2e_gemm_neg(const __bf16* __restrict__ eb,
                              const __bf16* __restrict__ cb,
                              const float*  __restrict__ bp,
                              float* __restrict__ negval) {
  const int gwave = (int)((blockIdx.x * blockDim.x + threadIdx.x) >> 5);
  const int lane  = threadIdx.x & 31;
  const int r0    = gwave * 16;          // first row (n*M + m index) of this tile
  const int l16   = lane & 15;
  const int kbase = (lane < 16) ? 0 : 8;

  const int spk    = r0 >> 5;            // owning speaker -> excluded column k == spk
  // spk is mathematically wave-uniform; readfirstlane puts the tile index in an SGPR
  // so the uniform half of the exclusion predicate stays scalar.
  const int exTile = __builtin_amdgcn_readfirstlane(spk >> 4);
  const bool exm   = (l16 == (spk & 15));   // per-lane half of the predicate

  // A panel: 16 rows x 128 K, register-resident for the whole column sweep.
  const __bf16* arow = eb + (size_t)(r0 + l16) * D_DIM;
  v16bf A0 = load_mat16(arow,  0 + kbase);
  v16bf A1 = load_mat16(arow, 32 + kbase);
  v16bf A2 = load_mat16(arow, 64 + kbase);
  v16bf A3 = load_mat16(arow, 96 + kbase);

  const float FINF = __builtin_huge_valf();
  float best[8];
#pragma unroll
  for (int r = 0; r < 8; r++) best[r] = -FINF;

  const __bf16* bbase = cb + (size_t)l16 * D_DIM;        // this lane's column of tile 0
  const __bf16* pf    = cb + (size_t)(48 + l16) * D_DIM; // prefetch cursor, 3 tiles ahead

  for (int tt = 0; tt < 32; tt++) {
    const int ta = 2 * tt, tb = 2 * tt + 1;
    const __bf16* bca = bbase + (size_t)(ta * 16) * D_DIM;
    const __bf16* bcb = bca + (size_t)(16 * D_DIM);

    // Pointer-carried prefetch of the B panel (gfx1250 global_prefetch_b8);
    // small overrun past cb lands inside the same workspace (harmless).
    __builtin_prefetch(pf, 0, 3);
    pf += 2 * 16 * D_DIM;

    v16bf Ba0 = load_mat16(bca,  0 + kbase);
    v16bf Ba1 = load_mat16(bca, 32 + kbase);
    v16bf Ba2 = load_mat16(bca, 64 + kbase);
    v16bf Ba3 = load_mat16(bca, 96 + kbase);
    v16bf Bb0 = load_mat16(bcb,  0 + kbase);
    v16bf Bb1 = load_mat16(bcb, 32 + kbase);
    v16bf Bb2 = load_mat16(bcb, 64 + kbase);
    v16bf Bb3 = load_mat16(bcb, 96 + kbase);

    v8f accA = {};
    v8f accB = {};
    accA = __builtin_amdgcn_wmma_f32_16x16x32_bf16(false, A0, false, Ba0, (short)0, accA, false, false);
    accB = __builtin_amdgcn_wmma_f32_16x16x32_bf16(false, A0, false, Bb0, (short)0, accB, false, false);
    accA = __builtin_amdgcn_wmma_f32_16x16x32_bf16(false, A1, false, Ba1, (short)0, accA, false, false);
    accB = __builtin_amdgcn_wmma_f32_16x16x32_bf16(false, A1, false, Bb1, (short)0, accB, false, false);
    accA = __builtin_amdgcn_wmma_f32_16x16x32_bf16(false, A2, false, Ba2, (short)0, accA, false, false);
    accB = __builtin_amdgcn_wmma_f32_16x16x32_bf16(false, A2, false, Bb2, (short)0, accB, false, false);
    accA = __builtin_amdgcn_wmma_f32_16x16x32_bf16(false, A3, false, Ba3, (short)0, accA, false, false);
    accB = __builtin_amdgcn_wmma_f32_16x16x32_bf16(false, A3, false, Bb3, (short)0, accB, false, false);

    // Branch-free leave-one-out masking: no control flow, no phi copies.
    const bool hitA = (ta == exTile) && exm;
    const bool hitB = (tb == exTile) && exm;
#pragma unroll
    for (int r = 0; r < 8; r++) {
      float va = hitA ? -FINF : accA[r];
      float vb = hitB ? -FINF : accB[r];
      best[r] = fmaxf(best[r], fmaxf(va, vb));
    }
  }

  // Reduce over the 16 columns (lanes) of each half-wave.
#pragma unroll
  for (int o = 1; o < 16; o <<= 1) {
#pragma unroll
    for (int r = 0; r < 8; r++)
      best[r] = fmaxf(best[r], __shfl_xor(best[r], o, 32));
  }

  if (l16 == 0) {
    const float bb = bp[0];
    const int roff = (lane < 16) ? 0 : 8;  // lane<16: rows r0..r0+7; lane>=16: rows r0+8..r0+15
#pragma unroll
    for (int r = 0; r < 8; r++)
      negval[r0 + roff + r] = sigmoidf_(best[r] + bb);   // best already holds w*cos
  }
}

// Kernel 3: deterministic single-block sum of pos + neg terms.
__global__ void ge2e_reduce(const float* __restrict__ posval,
                            const float* __restrict__ negval,
                            float* __restrict__ out) {
  __shared__ float sc[32];
  float s = 0.f;
  for (int i = threadIdx.x; i < NROWS; i += blockDim.x)
    s += posval[i] + negval[i];
#pragma unroll
  for (int o = 16; o > 0; o >>= 1) s += __shfl_xor(s, o, 32);
  const int wid = threadIdx.x >> 5;
  if ((threadIdx.x & 31) == 0) sc[wid] = s;
  __syncthreads();
  if (wid == 0) {
    float v = sc[threadIdx.x & 31];          // blockDim 1024 -> exactly 32 wave partials
#pragma unroll
    for (int o = 16; o > 0; o >>= 1) v += __shfl_xor(v, o, 32);
    if (threadIdx.x == 0) out[0] = v;
  }
}

extern "C" void kernel_launch(void* const* d_in, const int* in_sizes, int n_in,
                              void* d_out, int out_size, void* d_ws, size_t ws_size,
                              hipStream_t stream) {
  const float* emb = (const float*)d_in[0];
  const float* wp  = (const float*)d_in[1];
  const float* bp  = (const float*)d_in[2];
  float* out = (float*)d_out;

  char* ws = (char*)d_ws;
  __bf16* eb = (__bf16*)ws;                                          // 32768*128*2 = 8 MB
  __bf16* cb = (__bf16*)(ws + (size_t)NROWS * D_DIM * 2);            // 1024*128*2  = 256 KB
  float* posval = (float*)(ws + (size_t)NROWS * D_DIM * 2
                              + (size_t)N_SPK * D_DIM * 2);          // 128 KB
  float* negval = posval + NROWS;                                    // 128 KB

  ge2e_prep<<<N_SPK, 128, 0, stream>>>(emb, wp, bp, eb, cb, posval);
  // 32768/16 = 2048 waves, 8 waves (256 threads) per block -> 256 blocks
  ge2e_gemm_neg<<<256, 256, 0, stream>>>(eb, cb, bp, negval);
  ge2e_reduce<<<1, 1024, 0, stream>>>(posval, negval, out);
}
